// NestedAttention_36747740185073
// MI455X (gfx1250) — compile-verified, matrix operand order checked by reading, not run on hardware
//
#include <hip/hip_runtime.h>

// ---------------------------------------------------------------------------
// NestedAttention on gfx1250: bf16 WMMA for all matrix stages, fp32 softmax,
// fp32 accumulate. GEMMs stage A/B tiles into LDS via the Tensor Data Mover
// (tensor_load_to_lds, double-buffered on TENSORcnt) when available.
// ---------------------------------------------------------------------------

typedef __bf16 bf16_t;
typedef __attribute__((ext_vector_type(8)))  __bf16 v8bf;
typedef __attribute__((ext_vector_type(16))) __bf16 v16bf;
typedef __attribute__((ext_vector_type(8)))  float  v8f;
typedef __attribute__((ext_vector_type(4)))  unsigned int v4u;
typedef __attribute__((ext_vector_type(4)))  int v4i;
typedef __attribute__((ext_vector_type(8)))  int v8i;

#define BB   8
#define NN   1024
#define DD   1024
#define HH   16
#define HD   64
#define MTOK (BB * NN)   // 8192 tokens

#define HAVE_TDM __has_builtin(__builtin_amdgcn_tensor_load_to_lds)

__device__ __forceinline__ v8f wmma_bf16(v16bf a, v16bf b, v8f c) {
  return __builtin_amdgcn_wmma_f32_16x16x32_bf16(false, a, false, b, (short)0, c,
                                                 false, false);
}

// A/B fragment loader for V_WMMA_F32_16X16X32_BF16 (ISA 7.12.2):
// lane = g*16 + r holds row r; elements e<8 -> K=g*8+e, e>=8 -> K=16+g*8+(e-8).
__device__ __forceinline__ v16bf load_frag(const bf16_t* p, int ld, int row0, int k0) {
  const int lane = threadIdx.x & 31;
  const int g = lane >> 4;
  const int r = lane & 15;
  const bf16_t* base = p + (size_t)(row0 + r) * (size_t)ld + k0 + g * 8;
  v8bf lo = *(const v8bf*)(base);
  v8bf hi = *(const v8bf*)(base + 16);
  v16bf f;
#pragma unroll
  for (int i = 0; i < 8; ++i) { f[i] = lo[i]; f[8 + i] = hi[i]; }
  return f;
}

// --------------------------- TDM tile loader -------------------------------
#if HAVE_TDM
__device__ __forceinline__ unsigned lds_byte_addr(const void* p) {
  // local->generic addrspace cast is {aperture_hi, lds_offset}; low 32 bits
  // are the LDS byte address the D# wants.
  return (unsigned)(size_t)p;
}

// DMA a 64-row x 32-col bf16 tile (row stride 1024 elems) into LDS [64][32].
// D# per CDNA5 ISA 8.3/8.4: group0 = count/lds/global/type, group1 = dims.
__device__ __forceinline__ void tdm_load_tile_64x32(const bf16_t* gsrc,
                                                    unsigned ldsAddr,
                                                    unsigned tensorRows) {
  unsigned long long ga = (unsigned long long)(size_t)gsrc;
  v4u g0;
  g0[0] = 1u;                                            // count=1 user D#
  g0[1] = ldsAddr;                                       // lds_addr (bytes)
  g0[2] = (unsigned)(ga & 0xFFFFFFFFull);                // global_addr[31:0]
  g0[3] = (unsigned)((ga >> 32) & 0x01FFFFFFull)         // global_addr[56:32]
        | (2u << 30);                                    // type=2 (image)
  v8i g1;
  g1[0] = (int)(1u << 16);                               // data_size=1 (2B)
  g1[1] = (int)((DD & 0xFFFFu) << 16);                   // tensor_dim0 lo16
  g1[2] = (int)(((DD >> 16) & 0xFFFFu)                   // tensor_dim0 hi16
        | ((tensorRows & 0xFFFFu) << 16));               // tensor_dim1 lo16
  g1[3] = (int)(((tensorRows >> 16) & 0xFFFFu)           // tensor_dim1 hi16
        | (32u << 16));                                  // tile_dim0 = 32
  g1[4] = (int)64u;                                      // tile_dim1 = 64
  g1[5] = (int)DD;                                       // dim0_stride lo32
  g1[6] = 0;                                             // stride hi / dim1_stride
  g1[7] = 0;
  v4i z4 = {0, 0, 0, 0};
#if __clang_major__ >= 23
  v8i z8 = {0, 0, 0, 0, 0, 0, 0, 0};
  __builtin_amdgcn_tensor_load_to_lds(g0, g1, z4, z4, z8, 0);
#else
  __builtin_amdgcn_tensor_load_to_lds(g0, g1, z4, z4, 0);
#endif
}
#endif  // HAVE_TDM

// Shared 64x64 block mainloop: C[2][2] wave accumulators, K = DD in steps of 32.
__device__ __forceinline__ void mma_block_64x64(
    const bf16_t* __restrict__ A, const bf16_t* __restrict__ B,
    int mblk, int nblk, unsigned tensorRowsB,
    bf16_t (&As)[2][64][32], bf16_t (&Bs)[2][64][32], v8f acc[2][2]) {
  const int w = threadIdx.x >> 5;
  const int m0l = (w >> 1) * 32;
  const int n0l = (w & 1) * 32;
#if HAVE_TDM
  if (w == 0) {  // one wave issues the DMAs (TDM ignores EXEC; wave-level op)
    tdm_load_tile_64x32(A + (size_t)mblk * DD, lds_byte_addr(&As[0][0][0]), MTOK);
    tdm_load_tile_64x32(B + (size_t)nblk * DD, lds_byte_addr(&Bs[0][0][0]), tensorRowsB);
  }
  for (int i = 0; i < DD / 32; ++i) {
    const int cur = i & 1, nxt = cur ^ 1;
    if (w == 0) {
      if (i + 1 < DD / 32) {
        const int k = (i + 1) * 32;
        tdm_load_tile_64x32(A + (size_t)mblk * DD + k,
                            lds_byte_addr(&As[nxt][0][0]), MTOK);
        tdm_load_tile_64x32(B + (size_t)nblk * DD + k,
                            lds_byte_addr(&Bs[nxt][0][0]), tensorRowsB);
        // TDM ops retire in order per wave: <=2 outstanding means the two
        // current-buffer tiles have landed; only next-buffer DMAs in flight.
        __builtin_amdgcn_s_wait_tensorcnt(2);
      } else {
        __builtin_amdgcn_s_wait_tensorcnt(0);
      }
    }
    __syncthreads();
    v16bf a0 = load_frag(&As[cur][0][0], 32, m0l, 0);
    v16bf a1 = load_frag(&As[cur][0][0], 32, m0l + 16, 0);
    v16bf b0 = load_frag(&Bs[cur][0][0], 32, n0l, 0);
    v16bf b1 = load_frag(&Bs[cur][0][0], 32, n0l + 16, 0);
    acc[0][0] = wmma_bf16(a0, b0, acc[0][0]);
    acc[0][1] = wmma_bf16(a0, b1, acc[0][1]);
    acc[1][0] = wmma_bf16(a1, b0, acc[1][0]);
    acc[1][1] = wmma_bf16(a1, b1, acc[1][1]);
    __syncthreads();  // all waves done with buf[cur] before it is overwritten
  }
#else
  (void)As; (void)Bs;
  for (int k = 0; k < DD; k += 32) {
    v16bf a0 = load_frag(A, DD, mblk + m0l, k);
    v16bf a1 = load_frag(A, DD, mblk + m0l + 16, k);
    v16bf b0 = load_frag(B, DD, nblk + n0l, k);
    v16bf b1 = load_frag(B, DD, nblk + n0l + 16, k);
    acc[0][0] = wmma_bf16(a0, b0, acc[0][0]);
    acc[0][1] = wmma_bf16(a0, b1, acc[0][1]);
    acc[1][0] = wmma_bf16(a1, b0, acc[1][0]);
    acc[1][1] = wmma_bf16(a1, b1, acc[1][1]);
  }
#endif
}

// --------------------------- conversion kernels ----------------------------

__global__ __launch_bounds__(256) void cvt_x_kernel(const float* __restrict__ x,
                                                    const int* __restrict__ em,
                                                    bf16_t* __restrict__ xm) {
  size_t i = (size_t)blockIdx.x * blockDim.x + threadIdx.x;
  int d = (int)(i & (DD - 1));
  size_t tok = i >> 10;
  int cap = DD >> (3 - em[tok]);   // caps_table[e] = D >> (E-1-e)
  float v = x[i];
  xm[i] = (bf16_t)((d < cap) ? v : 0.0f);
}

__global__ __launch_bounds__(256) void cvt_wt_kernel(const float* __restrict__ w,
                                                     bf16_t* __restrict__ wt,
                                                     int K, int Nc) {
  size_t i = (size_t)blockIdx.x * blockDim.x + threadIdx.x;  // over Nc*K
  int k = (int)(i % K);
  int n = (int)(i / K);
  wt[i] = (bf16_t)w[(size_t)k * Nc + n];
}

// ------------------------------ QKV GEMM -----------------------------------
__global__ __launch_bounds__(128) void gemm_qkv_kernel(
    const bf16_t* __restrict__ Xm, const bf16_t* __restrict__ Wt,
    bf16_t* __restrict__ Qb, bf16_t* __restrict__ Kb, bf16_t* __restrict__ Vt) {
  __shared__ __align__(16) bf16_t As[2][64][32];
  __shared__ __align__(16) bf16_t Bs[2][64][32];
  const int lane = threadIdx.x & 31;
  const int w = threadIdx.x >> 5;
  const int mblk = blockIdx.y * 64;
  const int nblk = blockIdx.x * 64;
  v8f acc[2][2] = {};
  mma_block_64x64(Xm, Wt, mblk, nblk, 3 * DD, As, Bs, acc);

  const int m0 = mblk + (w >> 1) * 32;
  const int n0 = nblk + (w & 1) * 32;
#pragma unroll
  for (int i = 0; i < 2; ++i)
#pragma unroll
    for (int j = 0; j < 2; ++j)
#pragma unroll
      for (int r = 0; r < 8; ++r) {
        int row = m0 + i * 16 + ((lane < 16) ? r : r + 8);
        int col = n0 + j * 16 + (lane & 15);
        int s = col >> 10;           // 0=q 1=k 2=v
        int h = (col & 1023) >> 6;
        int d = col & 63;
        int b = row >> 10;
        int n = row & 1023;
        bf16_t v = (bf16_t)acc[i][j][r];
        size_t head = (size_t)b * HH + h;
        if (s == 0)      Qb[(head * NN + n) * HD + d] = v;
        else if (s == 1) Kb[(head * NN + n) * HD + d] = v;
        else             Vt[(head * HD + d) * NN + n] = v;   // transposed V
      }
}

// ------------------------------ attention ----------------------------------
__global__ __launch_bounds__(128) void attn_kernel(
    const bf16_t* __restrict__ Qb, const bf16_t* __restrict__ Kb,
    const bf16_t* __restrict__ Vt, bf16_t* __restrict__ Xa) {
  __shared__ __align__(16) bf16_t Plds[4][16][32];   // per-wave P tile
  const int lane = threadIdx.x & 31;
  const int w = threadIdx.x >> 5;
  const int gw = blockIdx.x * 4 + w;
  const int qt = gw & 63;           // N/16 = 64 q-tiles
  const int bh = gw >> 6;           // b*H + h
  const int b = bh >> 4;
  const int h = bh & 15;
  const int q0 = qt * 16;
  const bf16_t* Qh = Qb + (size_t)bh * NN * HD;
  const bf16_t* Kh = Kb + (size_t)bh * NN * HD;
  const bf16_t* Vh = Vt + (size_t)bh * HD * NN;

  v16bf qa0 = load_frag(Qh, HD, q0, 0);
  v16bf qa1 = load_frag(Qh, HD, q0, 32);
  float mrow[8], lrow[8];
#pragma unroll
  for (int r = 0; r < 8; ++r) { mrow[r] = -1e30f; lrow[r] = 0.0f; }
  v8f acco[4] = {};
  const float scale = 0.125f;   // hd^-0.5

  for (int kc = 0; kc < NN; kc += 32) {
    v8f s0 = {}, s1 = {};
    s0 = wmma_bf16(qa0, load_frag(Kh, HD, kc, 0), s0);
    s0 = wmma_bf16(qa1, load_frag(Kh, HD, kc, 32), s0);
    s1 = wmma_bf16(qa0, load_frag(Kh, HD, kc + 16, 0), s1);
    s1 = wmma_bf16(qa1, load_frag(Kh, HD, kc + 16, 32), s1);

    float tmax[8], tsum[8], alpha[8];
#pragma unroll
    for (int r = 0; r < 8; ++r) {
      float a = s0[r] * scale, c = s1[r] * scale;
      s0[r] = a; s1[r] = c;
      tmax[r] = fmaxf(a, c);
    }
#pragma unroll
    for (int off = 8; off >= 1; off >>= 1)
#pragma unroll
      for (int r = 0; r < 8; ++r)
        tmax[r] = fmaxf(tmax[r], __shfl_xor(tmax[r], off, 32));
#pragma unroll
    for (int r = 0; r < 8; ++r) {
      float mn = fmaxf(mrow[r], tmax[r]);
      alpha[r] = __expf(mrow[r] - mn);
      mrow[r] = mn;
      s0[r] = __expf(s0[r] - mn);
      s1[r] = __expf(s1[r] - mn);
      tsum[r] = s0[r] + s1[r];
    }
#pragma unroll
    for (int off = 8; off >= 1; off >>= 1)
#pragma unroll
      for (int r = 0; r < 8; ++r)
        tsum[r] += __shfl_xor(tsum[r], off, 32);
#pragma unroll
    for (int r = 0; r < 8; ++r) lrow[r] = lrow[r] * alpha[r] + tsum[r];
#pragma unroll
    for (int t = 0; t < 4; ++t)
#pragma unroll
      for (int r = 0; r < 8; ++r) acco[t][r] *= alpha[r];

    {  // C-layout -> A-layout via per-wave LDS tile (wave-private, DS in-order)
      int key = lane & 15;
#pragma unroll
      for (int r = 0; r < 8; ++r) {
        int row = (lane < 16) ? r : r + 8;
        Plds[w][row][key]      = (bf16_t)s0[r];
        Plds[w][row][key + 16] = (bf16_t)s1[r];
      }
    }
    v16bf pa = load_frag(&Plds[w][0][0], 32, 0, 0);
#pragma unroll
    for (int t = 0; t < 4; ++t) {
      v16bf vb = load_frag(Vh, NN, t * 16, kc);   // Vt rows = hd dims
      acco[t] = wmma_bf16(pa, vb, acco[t]);
    }
  }

#pragma unroll
  for (int t = 0; t < 4; ++t)
#pragma unroll
    for (int r = 0; r < 8; ++r) {
      int row = q0 + ((lane < 16) ? r : r + 8);
      int col = h * HD + t * 16 + (lane & 15);
      float v = acco[t][r] / lrow[r];
      Xa[((size_t)b * NN + row) * DD + col] = (bf16_t)v;
    }
}

// ------------------------------ output GEMM --------------------------------
__global__ __launch_bounds__(128) void gemm_out_kernel(
    const bf16_t* __restrict__ Xa, const bf16_t* __restrict__ Wt,
    const float* __restrict__ bias, const int* __restrict__ emask,
    float* __restrict__ out) {
  __shared__ __align__(16) bf16_t As[2][64][32];
  __shared__ __align__(16) bf16_t Bs[2][64][32];
  const int lane = threadIdx.x & 31;
  const int w = threadIdx.x >> 5;
  const int mblk = blockIdx.y * 64;
  const int nblk = blockIdx.x * 64;
  v8f acc[2][2] = {};
  mma_block_64x64(Xa, Wt, mblk, nblk, DD, As, Bs, acc);

  const int m0 = mblk + (w >> 1) * 32;
  const int n0 = nblk + (w & 1) * 32;
#pragma unroll
  for (int i = 0; i < 2; ++i)
#pragma unroll
    for (int j = 0; j < 2; ++j)
#pragma unroll
      for (int r = 0; r < 8; ++r) {
        int row = m0 + i * 16 + ((lane < 16) ? r : r + 8);
        int col = n0 + j * 16 + (lane & 15);
        int cap = DD >> (3 - emask[row]);
        float v = acc[i][j][r] + bias[col];
        out[(size_t)row * DD + col] = (col < cap) ? v : 0.0f;
      }
}

// ------------------------------- launcher ----------------------------------
extern "C" void kernel_launch(void* const* d_in, const int* in_sizes, int n_in,
                              void* d_out, int out_size, void* d_ws, size_t ws_size,
                              hipStream_t stream) {
  const float* x    = (const float*)d_in[0];   // (8,1024,1024) f32
  const int*   em   = (const int*)d_in[1];     // (8,1024) i32
  const float* wqkv = (const float*)d_in[2];   // (1024,3072) f32
  const float* wprj = (const float*)d_in[3];   // (1024,1024) f32
  const float* bprj = (const float*)d_in[4];   // (1024,) f32
  float* out = (float*)d_out;                  // (8,1024,1024) f32

  char* ws = (char*)d_ws;
  size_t off = 0;
  bf16_t* Xm  = (bf16_t*)(ws + off); off += (size_t)MTOK * DD * 2;      // 16 MB
  bf16_t* WqT = (bf16_t*)(ws + off); off += (size_t)(3 * DD) * DD * 2;  //  6 MB
  bf16_t* WpT = (bf16_t*)(ws + off); off += (size_t)DD * DD * 2;        //  2 MB
  bf16_t* Qb  = (bf16_t*)(ws + off); off += (size_t)MTOK * DD * 2;      // 16 MB
  bf16_t* Kb  = (bf16_t*)(ws + off); off += (size_t)MTOK * DD * 2;      // 16 MB
  bf16_t* Vt  = (bf16_t*)(ws + off); off += (size_t)MTOK * DD * 2;      // 16 MB
  bf16_t* Xa  = (bf16_t*)(ws + off); off += (size_t)MTOK * DD * 2;      // 16 MB
  (void)ws_size; (void)in_sizes; (void)n_in; (void)out_size;

  cvt_x_kernel<<<(MTOK * DD) / 256, 256, 0, stream>>>(x, em, Xm);
  cvt_wt_kernel<<<(3 * DD * DD) / 256, 256, 0, stream>>>(wqkv, WqT, DD, 3 * DD);
  cvt_wt_kernel<<<(DD * DD) / 256, 256, 0, stream>>>(wprj, WpT, DD, DD);
  gemm_qkv_kernel<<<dim3((3 * DD) / 64, MTOK / 64), 128, 0, stream>>>(Xm, WqT, Qb, Kb, Vt);
  attn_kernel<<<(BB * HH * (NN / 16)) / 4, 128, 0, stream>>>(Qb, Kb, Vt, Xa);
  gemm_out_kernel<<<dim3(DD / 64, MTOK / 64), 128, 0, stream>>>(Xa, WpT, bprj, em, out);
}
